// AttentionHead_82652350644542
// MI455X (gfx1250) — matrix operand order repaired and checked by reading, not compiled, not run on hardware
//
#include <hip/hip_runtime.h>
#include <hip/hip_bf16.h>

#define E_DIM 2048
#define D_DIM 128
#define BATCH 4
#define SEQ   2048
#define BT    (BATCH * SEQ)
#define KSPLIT 8                 // split-K factor for the OV GEMM
#define KCHUNK (SEQ / KSPLIT)    // 256 K-elements per split
#define RCHUNK (SEQ / 8)         // 256 rows per softmax reduction chunk

typedef __attribute__((ext_vector_type(16))) __bf16 v16bf;
typedef __attribute__((ext_vector_type(8)))  float  v8f;

__device__ __forceinline__ v8f wmma_bf16(v16bf a, v16bf b, v8f c) {
    // D = A(16x32 bf16) * B(32x16 bf16) + C(16x16 f32)
    return __builtin_amdgcn_wmma_f32_16x16x32_bf16(
        /*neg_a=*/false, a, /*neg_b=*/false, b,
        /*c_mod=*/(short)0, c, /*reuse_a=*/false, /*reuse_b=*/false);
}

// -------------------------------------------------------------------------
// Kernel 1: Q/K/V projections.  out[r, n] = sum_k x[r,k] * W[k,n]
// Block = 256 threads (8 waves), 128 rows x 128 cols per block.
// W chunk (32 x 128) staged in LDS pre-swizzled to the WMMA B-fragment
// layout so each lane reads its 16 bf16 (32B) contiguously.
// -------------------------------------------------------------------------
__global__ __launch_bounds__(256)
void ah_proj_kernel(const float* __restrict__ x,
                    const float* __restrict__ Wq,
                    const float* __restrict__ Wk,
                    const float* __restrict__ Wv,
                    __bf16* __restrict__ qb,
                    __bf16* __restrict__ kb,
                    __bf16* __restrict__ vb) {
    const float* W   = (blockIdx.y == 0) ? Wq : (blockIdx.y == 1) ? Wk : Wv;
    __bf16*      out = (blockIdx.y == 0) ? qb : (blockIdx.y == 1) ? kb : vb;

    __shared__ __align__(32) __bf16 ldsB[32 * D_DIM]; // 8 KB, [ntile][lane][e]

    const int tid  = threadIdx.x;
    const int wave = tid >> 5;
    const int lane = tid & 31;
    const int hi   = (lane & 16) ? 1 : 0;
    const int l15  = lane & 15;

    const int    rowg = blockIdx.x * 128 + wave * 16 + l15; // A-frag row of this lane
    const float* xrow = x + (size_t)rowg * E_DIM;

    v8f acc[8] = {};

    for (int kk = 0; kk < E_DIM; kk += 32) {
        // Stage 32x128 W chunk into LDS as bf16 in B-frag order:
        // blob[nt][lane][e] = W[kk + ((lane<16)?0:8) + (e<8 ? e : e+8)][nt*16 + (lane&15)]
        for (int f = tid; f < 32 * D_DIM; f += 256) {
            int nt   = f >> 9;
            int rem  = f & 511;
            int ln   = rem >> 4;
            int e    = rem & 15;
            int kloc = ((ln & 16) ? 8 : 0) + (e < 8 ? e : e + 8);
            int n    = nt * 16 + (ln & 15);
            ldsB[f] = (__bf16)W[(size_t)(kk + kloc) * D_DIM + n];
        }
        __syncthreads();

        const float* xp = xrow + kk + (hi ? 8 : 0);
        if (kk + 32 < E_DIM)
            __builtin_prefetch(xp + 32, 0, 0);   // global_prefetch_b8: next x chunk

        // A fragment (16-bit A 16x32 layout): K = base+e (e<8), base+16+(e-8).
        v16bf a;
        #pragma unroll
        for (int e = 0; e < 8; ++e)  a[e] = (__bf16)xp[e];
        #pragma unroll
        for (int e = 8; e < 16; ++e) a[e] = (__bf16)xp[e + 8];

        #pragma unroll
        for (int nt = 0; nt < 8; ++nt) {
            v16bf bf = *(const v16bf*)&ldsB[nt * 512 + lane * 16];
            acc[nt] = wmma_bf16(a, bf, acc[nt]);
        }
        __syncthreads();
    }

    // C/D layout: VGPR r holds M = r (lanes 0-15) / 8+r (lanes 16-31), N = lane&15.
    const int rowo = blockIdx.x * 128 + wave * 16 + hi * 8;
    #pragma unroll
    for (int nt = 0; nt < 8; ++nt) {
        int col = nt * 16 + l15;
        #pragma unroll
        for (int r = 0; r < 8; ++r)
            out[(size_t)(rowo + r) * D_DIM + col] = (__bf16)acc[nt][r];
    }
}

// -------------------------------------------------------------------------
// Kernel 2: S[b,t,s] = (q[b,t,:] . k[b,s,:]) / sqrt(D), causal mask (s>t -> -1e30)
// B-fragments come straight from K rows (contiguous 32B loads).  Blocks fully
// above the diagonal skip the GEMM and just write the mask value.
// -------------------------------------------------------------------------
__global__ __launch_bounds__(256)
void ah_scores_kernel(const __bf16* __restrict__ qb,
                      const __bf16* __restrict__ kb,
                      float* __restrict__ S) {
    const int b    = blockIdx.z;
    const int tr   = blockIdx.x * 128;   // query rows
    const int tc   = blockIdx.y * 128;   // key cols
    const int tid  = threadIdx.x;
    const int wave = tid >> 5;
    const int lane = tid & 31;
    const int hi   = (lane & 16) ? 1 : 0;
    const int l15  = lane & 15;
    const float scale = 0.088388347648318447f; // 1/sqrt(128)

    if (blockIdx.y > blockIdx.x) {           // fully masked tile
        #pragma unroll
        for (int nt = 0; nt < 8; ++nt) {
            int s = tc + nt * 16 + l15;
            #pragma unroll
            for (int r = 0; r < 8; ++r) {
                int t = tr + wave * 16 + hi * 8 + r;
                S[((size_t)b * SEQ + t) * SEQ + s] = -1.0e30f;
            }
        }
        return;
    }

    v8f acc[8] = {};
    const __bf16* qrow  = qb + ((size_t)b * SEQ + tr + wave * 16 + l15) * D_DIM;
    const int     kbase = hi ? 16 : 0;

    #pragma unroll
    for (int kk = 0; kk < D_DIM; kk += 32) {
        v16bf a;
        const __bf16* ap = qrow + kk + (hi ? 16 : 0);
        #pragma unroll
        for (int e = 0; e < 8; ++e)  a[e] = ap[e];
        #pragma unroll
        for (int e = 8; e < 16; ++e) a[e] = ap[e];

        #pragma unroll
        for (int nt = 0; nt < 8; ++nt) {
            const __bf16* kp = kb + ((size_t)b * SEQ + tc + nt * 16 + l15) * D_DIM + kk + kbase;
            v16bf bf = *(const v16bf*)kp;   // 32B aligned, contiguous K=kbase..kbase+15
            acc[nt] = wmma_bf16(a, bf, acc[nt]);
        }
    }

    #pragma unroll
    for (int nt = 0; nt < 8; ++nt) {
        int s = tc + nt * 16 + l15;
        #pragma unroll
        for (int r = 0; r < 8; ++r) {
            int t = tr + wave * 16 + hi * 8 + r;
            float v = acc[nt][r] * scale;
            if (s > t) v = -1.0e30f;
            S[((size_t)b * SEQ + t) * SEQ + s] = v;
        }
    }
}

// -------------------------------------------------------------------------
// Kernel 3a: partial column stats.  Softmax is over axis=1 (query/row axis),
// so reduce down each column; split the 2048 rows into 8 chunks for
// parallelism.  Adjacent threads -> adjacent columns (coalesced row reads).
// Writes online-softmax (m, d) pairs per (column, chunk).
// -------------------------------------------------------------------------
__global__ __launch_bounds__(256)
void ah_colstats1_kernel(const float* __restrict__ S,
                         float* __restrict__ pmM, float* __restrict__ pmD) {
    const int b  = blockIdx.z;
    const int rc = blockIdx.y;
    const int c  = blockIdx.x * 256 + threadIdx.x;
    const float* col = S + ((size_t)b * SEQ + rc * RCHUNK) * SEQ + c;

    float m = -1.0e30f, d = 0.0f;
    for (int t = 0; t < RCHUNK; ++t) {
        float xv = col[(size_t)t * SEQ];
        float mn = fmaxf(m, xv);
        d = d * __expf(m - mn) + __expf(xv - mn);
        m = mn;
    }
    size_t idx = ((size_t)b * SEQ + c) * 8 + rc;
    pmM[idx] = m;
    pmD[idx] = d;
}

// -------------------------------------------------------------------------
// Kernel 3b: combine the 8 partial (m,d) pairs per column into final
// column max m and reciprocal denominator 1/d.
// -------------------------------------------------------------------------
__global__ __launch_bounds__(256)
void ah_colstats2_kernel(const float* __restrict__ pmM, const float* __restrict__ pmD,
                         float* __restrict__ mfin, float* __restrict__ invd) {
    const int b = blockIdx.y;
    const int c = blockIdx.x * 256 + threadIdx.x;
    size_t base = ((size_t)b * SEQ + c) * 8;

    float M = -1.0e30f;
    #pragma unroll
    for (int rc = 0; rc < 8; ++rc) M = fmaxf(M, pmM[base + rc]);
    float D = 0.0f;
    #pragma unroll
    for (int rc = 0; rc < 8; ++rc) D += pmD[base + rc] * __expf(pmM[base + rc] - M);

    mfin[(size_t)b * SEQ + c] = M;
    invd[(size_t)b * SEQ + c] = 1.0f / D;
}

// -------------------------------------------------------------------------
// Kernel 4: split-K OV GEMM.  Z[b,t,d] = sum_s exp(S[t,s]-m[s])*invd[s]*V[s,d].
// invd[s] is folded into the V rows while staging the swizzled B fragments
// into LDS; A fragments are exp(S - m) computed on the fly (m staged in LDS
// per 32-wide K chunk).  Each block handles 256 of the 2048 K values and
// writes an fp32 partial; grid = 16 x 8 x 4 = 512 blocks.
// -------------------------------------------------------------------------
__global__ __launch_bounds__(256)
void ah_ov_kernel(const float* __restrict__ S,
                  const __bf16* __restrict__ vb,
                  const float* __restrict__ mfin,
                  const float* __restrict__ invd,
                  float* __restrict__ Zp) {
    const int b    = blockIdx.z;
    const int spl  = blockIdx.y;
    const int tr   = blockIdx.x * 128;
    const int k0   = spl * KCHUNK;
    const int tid  = threadIdx.x;
    const int wave = tid >> 5;
    const int lane = tid & 31;
    const int hi   = (lane & 16) ? 1 : 0;
    const int l15  = lane & 15;

    __shared__ __align__(32) __bf16 ldsB[32 * D_DIM];  // invd-scaled V, B-frag order
    __shared__ float ldsM[32];                          // column max for this K chunk

    const float* srow = S + ((size_t)b * SEQ + tr + wave * 16 + l15) * SEQ;

    v8f acc[8] = {};

    for (int step = 0; step < KCHUNK; step += 32) {
        const int kk = k0 + step;

        // Stage invd-scaled V chunk (32 x 128) in B-frag order + m chunk.
        for (int f = tid; f < 32 * D_DIM; f += 256) {
            int nt   = f >> 9;
            int rem  = f & 511;
            int ln   = rem >> 4;
            int e    = rem & 15;
            int kloc = ((ln & 16) ? 8 : 0) + (e < 8 ? e : e + 8);
            int n    = nt * 16 + (ln & 15);
            float vv = (float)vb[((size_t)b * SEQ + kk + kloc) * D_DIM + n];
            ldsB[f] = (__bf16)(vv * invd[(size_t)b * SEQ + kk + kloc]);
        }
        if (tid < 32) ldsM[tid] = mfin[(size_t)b * SEQ + kk + tid];
        __syncthreads();

        // A fragment: exp(S[t, kk+k] - m[kk+k]) as bf16 (values in (0,1]).
        const int    base = hi ? 8 : 0;
        const float* ap   = srow + kk + base;
        const float* mp   = ldsM + base;
        v16bf a;
        #pragma unroll
        for (int e = 0; e < 8; ++e)  a[e] = (__bf16)__expf(ap[e]     - mp[e]);
        #pragma unroll
        for (int e = 8; e < 16; ++e) a[e] = (__bf16)__expf(ap[e + 8] - mp[e + 8]);

        #pragma unroll
        for (int nt = 0; nt < 8; ++nt) {
            v16bf bf = *(const v16bf*)&ldsB[nt * 512 + lane * 16];
            acc[nt] = wmma_bf16(a, bf, acc[nt]);
        }
        __syncthreads();
    }

    const int rowo = tr + wave * 16 + hi * 8;
    float* zp = Zp + (size_t)spl * BT * D_DIM;
    #pragma unroll
    for (int nt = 0; nt < 8; ++nt) {
        int col = nt * 16 + l15;
        #pragma unroll
        for (int r = 0; r < 8; ++r)
            zp[((size_t)b * SEQ + rowo + r) * D_DIM + col] = acc[nt][r];
    }
}

// -------------------------------------------------------------------------
// Kernel 5: deterministic split-K reduction into the fp32 output.
// -------------------------------------------------------------------------
__global__ __launch_bounds__(256)
void ah_reduce_kernel(const float* __restrict__ Zp, float* __restrict__ z) {
    const size_t i = (size_t)blockIdx.x * 256 + threadIdx.x;
    float s = 0.0f;
    #pragma unroll
    for (int p = 0; p < KSPLIT; ++p)
        s += Zp[(size_t)p * BT * D_DIM + i];
    z[i] = s;
}

// -------------------------------------------------------------------------
extern "C" void kernel_launch(void* const* d_in, const int* in_sizes, int n_in,
                              void* d_out, int out_size, void* d_ws, size_t ws_size,
                              hipStream_t stream) {
    const float* x  = (const float*)d_in[0];
    const float* Wq = (const float*)d_in[1];
    const float* Wk = (const float*)d_in[2];
    const float* Wv = (const float*)d_in[3];
    float* z = (float*)d_out;

    char* ws = (char*)d_ws;
    const size_t MB = 1024 * 1024;
    __bf16* qb   = (__bf16*)(ws + 0 * MB);              //  2 MB (8192 x 128 bf16)
    __bf16* kb   = (__bf16*)(ws + 2 * MB);              //  2 MB
    __bf16* vb   = (__bf16*)(ws + 4 * MB);              //  2 MB
    float*  S    = (float*)(ws + 6 * MB);               // 64 MB (4 x 2048 x 2048 f32)
    float*  pmM  = (float*)(ws + 70 * MB);              // 256 KB (B*T*8)
    float*  pmD  = (float*)(ws + 70 * MB + 262144);     // 256 KB
    float*  mfin = (float*)(ws + 70 * MB + 524288);     // 32 KB (B*T)
    float*  invd = (float*)(ws + 70 * MB + 557056);     // 32 KB
    float*  Zp   = (float*)(ws + 71 * MB);              // 32 MB (KSPLIT x B*T*128 f32)

    ah_proj_kernel<<<dim3(BT / 128, 3), 256, 0, stream>>>(x, Wq, Wk, Wv, qb, kb, vb);
    ah_scores_kernel<<<dim3(SEQ / 128, SEQ / 128, BATCH), 256, 0, stream>>>(qb, kb, S);
    ah_colstats1_kernel<<<dim3(SEQ / 256, 8, BATCH), 256, 0, stream>>>(S, pmM, pmD);
    ah_colstats2_kernel<<<dim3(SEQ / 256, BATCH), 256, 0, stream>>>(pmM, pmD, mfin, invd);
    ah_ov_kernel<<<dim3(SEQ / 128, KSPLIT, BATCH), 256, 0, stream>>>(S, vb, mfin, invd, Zp);
    ah_reduce_kernel<<<dim3((BT * D_DIM) / 256), 256, 0, stream>>>(Zp, z);
}
